// CGCNN_15178414424372
// MI455X (gfx1250) — compile-verified, matrix operand order checked by reading, not run on hardware
//
#include <hip/hip_runtime.h>
#include <hip/hip_bf16.h>

// ---------------------------------------------------------------------------
// CGCNN forward for MI455X (gfx1250, wave32, WMMA).
// Edge GEMM [E,cin]@[cin,cin] fused with edge-MLP-1, gather h[src], and
// atomic scatter into agg[dst]; all GEMMs via v_wmma_f32_16x16x32_bf16.
// All tile bounds checks removed: E = 32*18750, nodes padded to 50016 = 32*1563.
// 32-bit indexing throughout hot kernels -> saddr+scale_offset addressing.
// ---------------------------------------------------------------------------

typedef __attribute__((ext_vector_type(16))) __bf16 v16bf;
typedef __attribute__((ext_vector_type(8)))  float  v8f;

#define NNODES 50000
#define NPAD   50016   // nodes padded to multiple of 32 (zeroed tail)
#define NEDGES 600000
#define NGRAPH 256
#define HDIM   128

__device__ __forceinline__ float softplus_f(float x) {
    return (x > 20.0f) ? x : log1pf(__expf(x));
}

__device__ __forceinline__ unsigned pack_bf16x2(float a, float b) {
    unsigned short lo = __builtin_bit_cast(unsigned short, (__bf16)a);
    unsigned short hi = __builtin_bit_cast(unsigned short, (__bf16)b);
    return (unsigned)lo | ((unsigned)hi << 16);
}

// ---- fragment loader (ISA 7.12.2 16-bit A 16x32 layout; B mirrored) -------
// lane 0-15 : M/N = lane,    K = [0..7] in elems 0..7, K = [16..23] in 8..15
// lane 16-31: M/N = lane-16, K = [8..15] and [24..31]
__device__ __forceinline__ v16bf load_frag_rowK(const __bf16* row) {
    v16bf f;
#pragma unroll
    for (int j = 0; j < 8; ++j) f[j] = row[j];
#pragma unroll
    for (int j = 0; j < 8; ++j) f[8 + j] = row[16 + j];
    return f;
}

// out[c*K + k] = (bf16) W[k*Ncols + c]  (transpose -> column-major bf16)
__global__ void wconvT_kernel(const float* __restrict__ W, __bf16* __restrict__ out,
                              int K, int Ncols) {
    int idx = blockIdx.x * blockDim.x + threadIdx.x;
    if (idx < K * Ncols) {
        int k = idx / Ncols, c = idx % Ncols;
        out[c * K + k] = (__bf16)W[idx];
    }
}

__global__ void zero_f32_kernel(float* __restrict__ p, int n) {
    for (int i = blockIdx.x * blockDim.x + threadIdx.x; i < n; i += gridDim.x * blockDim.x)
        p[i] = 0.0f;
}

// h0[n, 0:64] = emb[x[n]]   (N*64 = 3.2M, grid exact)
__global__ void embed_kernel(const int* __restrict__ x, const float* __restrict__ emb,
                             float* __restrict__ h) {
    int idx = blockIdx.x * blockDim.x + threadIdx.x;
    int n = idx >> 6, k = idx & 63;
    h[idx] = emb[x[n] * 64 + k];
}

// ---------------------------------------------------------------------------
// Fused edge kernel, 32 edges per block, CIN*4 threads = 2*(CIN/16) waves.
// Wave = (row-tile rt in 0..1) x (col-tile ct in 0..CIN/16-1).
//   t   = softplus(edge_attr @ W1 + b1)   -> LDS bf16 [32, CIN] (packed b32)
//   ee  = t @ W2 + b2                     -> WMMA bf16/f32
//   agg[dst] += ee * h[src]               -> clause of gathers + atomics
// ---------------------------------------------------------------------------
template <int CIN>
__global__ __launch_bounds__(CIN * 4) void edge_conv_kernel(
    const float* __restrict__ edge_attr,  // [E,4]
    const int* __restrict__ src, const int* __restrict__ dst,
    const float* __restrict__ ew1,        // [4,CIN] row-major f32
    const float* __restrict__ eb1,        // [CIN]
    const __bf16* __restrict__ ew2T,      // [CIN cols][CIN K] col-major bf16
    const float* __restrict__ eb2,        // [CIN]
    const float* __restrict__ h,          // [NPAD, CIN]
    float* __restrict__ agg)              // [NPAD, CIN]
{
    constexpr int CT = CIN / 16;          // col tiles per block
    __shared__ float  ea_s[32][4];
    __shared__ int    src_s[32];
    __shared__ int    dst_s[32];
    __shared__ __bf16 tA[32][CIN];

    const int e0 = blockIdx.x * 32;
    const int tid = threadIdx.x;
    constexpr int NT = CIN * 4;

    if (tid < 128) ea_s[tid >> 2][tid & 3] = edge_attr[e0 * 4 + tid];
    else if (tid < 160) { int m = tid - 128; src_s[m] = src[e0 + m]; dst_s[m] = dst[e0 + m]; }
    __syncthreads();

    // t = softplus(ea @ W1 + b1): 16*CIN bf16x2 pairs, 4 per thread, b32 stores
    unsigned* tAu = (unsigned*)&tA[0][0];
#pragma unroll
    for (int it = 0; it < 4; ++it) {
        int pidx = tid + it * NT;         // 4*NT == 16*CIN pairs
        int base = pidx * 2;
        int m = base / CIN, k = base % CIN;
        float v0 = eb1[k], v1 = eb1[k + 1];
#pragma unroll
        for (int f = 0; f < 4; ++f) {
            float e = ea_s[m][f];
            v0 += e * ew1[f * CIN + k];
            v1 += e * ew1[f * CIN + k + 1];
        }
        tAu[pidx] = pack_bf16x2(softplus_f(v0), softplus_f(v1));
    }

    const int wave = tid >> 5;
    const int lane = tid & 31;
    const int half = lane >> 4;
    const int ln   = lane & 15;
    const int rt   = wave / CT;           // 0..1
    const int ct   = wave - rt * CT;      // 0..CT-1
    const int col  = ct * 16 + ln;

    __syncthreads();

    // prefetch the h[src] gather rows while WMMAs run (global_prefetch_b8)
#pragma unroll
    for (int r = 0; r < 8; ++r) {
        int ml = rt * 16 + r + 8 * half;
        __builtin_prefetch(&h[src_s[ml] * CIN + col], 0, 0);
    }

    v8f acc = {0.f, 0.f, 0.f, 0.f, 0.f, 0.f, 0.f, 0.f};
#pragma unroll
    for (int ks = 0; ks < CIN / 32; ++ks) {
        v16bf a = load_frag_rowK(&tA[rt * 16 + ln][ks * 32 + half * 8]);
        v16bf b = load_frag_rowK(&ew2T[(ct * 16 + ln) * CIN + ks * 32 + half * 8]);
        acc = __builtin_amdgcn_wmma_f32_16x16x32_bf16(false, a, false, b,
                                                      (short)0, acc, false, false);
    }

    const float bias = eb2[col];
    float msg[8];
    int   doff[8];
#pragma unroll
    for (int r = 0; r < 8; ++r) {
        int ml = rt * 16 + r + 8 * half;
        int s = src_s[ml];
        doff[r] = dst_s[ml] * CIN + col;
        msg[r] = (acc[r] + bias) * h[s * CIN + col];
    }
#pragma unroll
    for (int r = 0; r < 8; ++r)
        atomicAdd(&agg[doff[r]], msg[r]);
}

// ---------------------------------------------------------------------------
// Node MLP: h_out[n, 0:128] = softplus(agg[n, 0:CIN] @ nw + nb)
// 32 nodes per block (NPAD multiple of 32), 512 threads = 16 waves
// = 2 row-tiles x 8 col-tiles.
// ---------------------------------------------------------------------------
template <int CIN>
__global__ __launch_bounds__(512) void node_mlp_kernel(
    const float* __restrict__ agg,    // [NPAD, CIN]
    const __bf16* __restrict__ nwT,   // [128 cols][CIN K] col-major bf16
    const float* __restrict__ nb,     // [128]
    float* __restrict__ hout)         // [NPAD, 128]
{
    __shared__ __bf16 tA[32 * CIN];
    const int n0 = blockIdx.x * 32;

    // stage agg tile: float2 loads -> packed bf16x2 b32 LDS stores
    const float2* aggv = (const float2*)(agg + n0 * CIN);
    unsigned* tAu = (unsigned*)tA;
#pragma unroll
    for (int i = threadIdx.x; i < 16 * CIN; i += 512) {
        float2 v = aggv[i];
        tAu[i] = pack_bf16x2(v.x, v.y);
    }
    __syncthreads();

    const int wave = threadIdx.x >> 5;
    const int lane = threadIdx.x & 31;
    const int half = lane >> 4;
    const int ln   = lane & 15;
    const int rt   = wave >> 3;       // 0..1
    const int ct   = wave & 7;        // 0..7
    const int col  = ct * 16 + ln;

    v8f acc = {0.f, 0.f, 0.f, 0.f, 0.f, 0.f, 0.f, 0.f};
#pragma unroll
    for (int ks = 0; ks < CIN / 32; ++ks) {
        v16bf a = load_frag_rowK(&tA[(rt * 16 + ln) * CIN + ks * 32 + half * 8]);
        v16bf b = load_frag_rowK(&nwT[(ct * 16 + ln) * CIN + ks * 32 + half * 8]);
        acc = __builtin_amdgcn_wmma_f32_16x16x32_bf16(false, a, false, b,
                                                      (short)0, acc, false, false);
    }

    const float bias = nb[col];
#pragma unroll
    for (int r = 0; r < 8; ++r) {
        int node = n0 + rt * 16 + r + 8 * half;
        hout[node * HDIM + col] = softplus_f(acc[r] + bias);
    }
}

// ---------------------------------------------------------------------------
// pooling + readout (scalar f32; negligible cost)
// ---------------------------------------------------------------------------
__global__ void pool_sum_kernel(const float* __restrict__ h, const int* __restrict__ batch,
                                float* __restrict__ gsum) {
    int idx = blockIdx.x * blockDim.x + threadIdx.x;   // N*128, grid exact
    int n = idx >> 7, c = idx & 127;
    atomicAdd(&gsum[batch[n] * HDIM + c], h[idx]);
}

__global__ void pool_cnt_kernel(const int* __restrict__ batch, float* __restrict__ cnt) {
    int n = blockIdx.x * blockDim.x + threadIdx.x;
    if (n < NNODES) atomicAdd(&cnt[batch[n]], 1.0f);
}

__global__ void pool_div_kernel(const float* __restrict__ gsum, const float* __restrict__ cnt,
                                float* __restrict__ g) {
    int idx = blockIdx.x * blockDim.x + threadIdx.x;   // G*128, grid exact
    g[idx] = gsum[idx] / fmaxf(cnt[idx >> 7], 1.0f);
}

template <bool SP>
__global__ void dense_kernel(const float* __restrict__ in, const float* __restrict__ W,
                             const float* __restrict__ b, float* __restrict__ out,
                             int FI, int FO) {
    int r = blockIdx.x, c = threadIdx.x;
    if (c >= FO) return;
    float v = b[c];
    for (int k = 0; k < FI; ++k) v += in[r * FI + k] * W[k * FO + c];
    if (SP) v = softplus_f(v);
    out[r * FO + c] = v;
}

// ---------------------------------------------------------------------------
extern "C" void kernel_launch(void* const* d_in, const int* in_sizes, int n_in,
                              void* d_out, int out_size, void* d_ws, size_t ws_size,
                              hipStream_t stream) {
    (void)in_sizes; (void)n_in; (void)out_size; (void)ws_size;

    // Inputs (setup_inputs dict order; params flattened in JAX pytree order:
    //   convs[0..2]{eb1,eb2,ew1,ew2,nb,nw}, emb, pb1,pb2,pw1,pw2, rb1,rb2,rw1,rw2)
    const int*   x     = (const int*)d_in[0];
    const int*   ei    = (const int*)d_in[1];
    const float* eattr = (const float*)d_in[2];
    const int*   batch = (const int*)d_in[3];

    struct ConvP { const float *eb1, *eb2, *ew1, *ew2, *nb, *nw; };
    ConvP cv[3];
    int p = 4;
    for (int l = 0; l < 3; ++l) {
        cv[l].eb1 = (const float*)d_in[p++];
        cv[l].eb2 = (const float*)d_in[p++];
        cv[l].ew1 = (const float*)d_in[p++];
        cv[l].ew2 = (const float*)d_in[p++];
        cv[l].nb  = (const float*)d_in[p++];
        cv[l].nw  = (const float*)d_in[p++];
    }
    const float* emb = (const float*)d_in[p++];
    const float* pb1 = (const float*)d_in[p++];
    const float* pb2 = (const float*)d_in[p++];
    const float* pw1 = (const float*)d_in[p++];
    const float* pw2 = (const float*)d_in[p++];
    const float* rb1 = (const float*)d_in[p++];
    const float* rb2 = (const float*)d_in[p++];
    const float* rw1 = (const float*)d_in[p++];
    const float* rw2 = (const float*)d_in[p++];

    const int* src = ei;
    const int* dst = ei + NEDGES;

    // workspace bump allocator
    char* ws = (char*)d_ws;
    size_t off = 0;
    auto alloc = [&](size_t bytes) -> void* {
        off = (off + 255) & ~(size_t)255;
        void* q = ws + off;
        off += bytes;
        return q;
    };

    float* hA  = (float*)alloc((size_t)NPAD * HDIM * 4);
    float* hB  = (float*)alloc((size_t)NPAD * HDIM * 4);
    float* agg = (float*)alloc((size_t)NPAD * HDIM * 4);

    const int cins[3] = {64, 128, 128};
    __bf16* ew2T[3];
    __bf16* nwT[3];
    for (int l = 0; l < 3; ++l) {
        ew2T[l] = (__bf16*)alloc((size_t)cins[l] * cins[l] * 2);
        nwT[l]  = (__bf16*)alloc((size_t)cins[l] * HDIM * 2);
    }
    float* gsum  = (float*)alloc((size_t)NGRAPH * HDIM * 4);
    float* cnt   = (float*)alloc((size_t)NGRAPH * 4);
    float* gpool = (float*)alloc((size_t)NGRAPH * HDIM * 4);
    float* pbuf1 = (float*)alloc((size_t)NGRAPH * HDIM * 4);
    float* pbuf2 = (float*)alloc((size_t)NGRAPH * HDIM * 4);
    float* rbuf1 = (float*)alloc((size_t)NGRAPH * 64 * 4);

    // ---- one-time weight conversion (bf16, column-major) ----
    for (int l = 0; l < 3; ++l) {
        int ne = cins[l] * cins[l];
        wconvT_kernel<<<(ne + 255) / 256, 256, 0, stream>>>(cv[l].ew2, ew2T[l], cins[l], cins[l]);
        int nn = cins[l] * HDIM;
        wconvT_kernel<<<(nn + 255) / 256, 256, 0, stream>>>(cv[l].nw, nwT[l], cins[l], HDIM);
    }

    // ---- embedding (zero padded tail of hA first so layer-0 tail reads 0) ----
    zero_f32_kernel<<<4096, 256, 0, stream>>>(hA, NPAD * 64);
    embed_kernel<<<(NNODES * 64) / 256, 256, 0, stream>>>(x, emb, hA);

    const int eblocks = NEDGES / 32;   // 18750, exact
    const int nblocks = NPAD / 32;     // 1563,  exact

    // ---- layer 0 (cin = 64): hA[.,64] -> hB[.,128] ----
    zero_f32_kernel<<<4096, 256, 0, stream>>>(agg, NPAD * 64);
    edge_conv_kernel<64><<<eblocks, 256, 0, stream>>>(eattr, src, dst,
        cv[0].ew1, cv[0].eb1, ew2T[0], cv[0].eb2, hA, agg);
    node_mlp_kernel<64><<<nblocks, 512, 0, stream>>>(agg, nwT[0], cv[0].nb, hB);

    // ---- layer 1 (cin = 128): hB -> hA ----
    zero_f32_kernel<<<4096, 256, 0, stream>>>(agg, NPAD * HDIM);
    edge_conv_kernel<128><<<eblocks, 512, 0, stream>>>(eattr, src, dst,
        cv[1].ew1, cv[1].eb1, ew2T[1], cv[1].eb2, hB, agg);
    node_mlp_kernel<128><<<nblocks, 512, 0, stream>>>(agg, nwT[1], cv[1].nb, hA);

    // ---- layer 2 (cin = 128): hA -> hB ----
    zero_f32_kernel<<<4096, 256, 0, stream>>>(agg, NPAD * HDIM);
    edge_conv_kernel<128><<<eblocks, 512, 0, stream>>>(eattr, src, dst,
        cv[2].ew1, cv[2].eb1, ew2T[2], cv[2].eb2, hA, agg);
    node_mlp_kernel<128><<<nblocks, 512, 0, stream>>>(agg, nwT[2], cv[2].nb, hB);

    // ---- global mean pool ----
    zero_f32_kernel<<<128, 256, 0, stream>>>(gsum, NGRAPH * HDIM);
    zero_f32_kernel<<<1, 256, 0, stream>>>(cnt, NGRAPH);
    pool_sum_kernel<<<(NNODES * HDIM) / 256, 256, 0, stream>>>(hB, batch, gsum);
    pool_cnt_kernel<<<(NNODES + 255) / 256, 256, 0, stream>>>(batch, cnt);
    pool_div_kernel<<<(NGRAPH * HDIM) / 256, 256, 0, stream>>>(gsum, cnt, gpool);

    // ---- readout MLPs (tiny) ----
    dense_kernel<true ><<<NGRAPH, HDIM, 0, stream>>>(gpool, pw1, pb1, pbuf1, HDIM, HDIM);
    dense_kernel<false><<<NGRAPH, HDIM, 0, stream>>>(pbuf1, pw2, pb2, pbuf2, HDIM, HDIM);
    dense_kernel<true ><<<NGRAPH, 64,   0, stream>>>(pbuf2, rw1, rb1, rbuf1, HDIM, 64);
    dense_kernel<false><<<NGRAPH, 1,    0, stream>>>(rbuf1, rw2, rb2, (float*)d_out, 64, 1);
}
